// Multi_GRL_47682726920832
// MI455X (gfx1250) — matrix operand order repaired and checked by reading, not compiled, not run on hardware
//
#include <hip/hip_runtime.h>
#include <hip/hip_bf16.h>

// Problem constants (from reference)
#define NSUBJ 15
#define KSRC  14        // S - 1
#define BATCH 1024
#define CIN   2048
#define DH    512
#define NCLS  3
#define M2    2048      // rows of "total" per k: [outs ; t_outs]

typedef __attribute__((ext_vector_type(16))) __bf16 v16bf;
typedef __attribute__((ext_vector_type(8)))  float  v8f;

// ---------- helpers ----------
__device__ __forceinline__ unsigned short f2bf(float f) {
  unsigned u = __float_as_uint(f);
  u += 0x7FFFu + ((u >> 16) & 1u);          // round-to-nearest-even
  return (unsigned short)(u >> 16);
}
__device__ __forceinline__ float bf2f(unsigned short s) {
  return __uint_as_float(((unsigned)s) << 16);
}
__device__ __forceinline__ float wave_sum(float v) {
#pragma unroll
  for (int o = 16; o > 0; o >>= 1) v += __shfl_xor(v, o, 32);
  return v;
}
__device__ __forceinline__ float block_sum(float v) {
  __shared__ float red[8];
  const int lane = threadIdx.x & 31, wid = threadIdx.x >> 5;
  v = wave_sum(v);
  __syncthreads();
  if (lane == 0) red[wid] = v;
  __syncthreads();
  float t = 0.f;
  if (wid == 0) {
    t = (lane < ((int)blockDim.x >> 5)) ? red[lane] : 0.f;
    t = wave_sum(t);
  }
  return t;
}

// WMMA 16x16x32 bf16 fragment swizzles (ISA 7.12.2, 16-bit A 16x32 / dense B 32x16)
__device__ __forceinline__ int a_frag_idx(int mm, int kk) { // element (M=mm,K=kk) in 16x32 tile
  int lane = mm + (((kk >> 3) & 1) << 4);
  int j    = (kk & 7) + ((kk >> 4) << 3);
  return lane * 16 + j;
}
__device__ __forceinline__ int b_frag_idx(int kk, int nn) { // element (K=kk,N=nn) in 32x16 tile
  int lane = nn + (((kk >> 3) & 1) << 4);
  int j    = (kk & 7) + ((kk >> 4) << 3);
  return lane * 16 + j;
}

// ---------- zero accumulators ----------
__global__ void k_zero(float* tprob, float* colsum, float* mmd_total,
                       float* disc_acc, float* cls_acc, int* pred) {
  int t = blockIdx.x * 256 + threadIdx.x;
  if (t < BATCH * NCLS) tprob[t] = 0.f;
  if (t < KSRC * DH)    colsum[t] = 0.f;
  if (t == 0) { *mmd_total = 0.f; *disc_acc = 0.f; *cls_acc = 0.f; *pred = 0; }
}

// ---------- layer-1 GEMM: total_k = [src_k;tgt] @ W_k^T + b_k  (bf16x3 WMMA) ----------
__global__ __launch_bounds__(256) void k_gemm1(
    const float* __restrict__ x, const float* __restrict__ W_add,
    const float* __restrict__ b_add, const int* __restrict__ subjp,
    unsigned short* __restrict__ tot_hi, unsigned short* __restrict__ tot_lo) {
  __shared__ unsigned short As[2][128 * 32];   // hi/lo planes, fragment-ordered
  __shared__ unsigned short Bs[2][32 * 64];
  const int tid = threadIdx.x, lane = tid & 31, wid = tid >> 5;
  const int waveM = wid >> 1, waveN = wid & 1;
  const int k = blockIdx.z;
  const int subj = subjp[0];
  const int ksub = k + (k >= subj);
  const int m0 = blockIdx.y * 128;
  const int n0 = blockIdx.x * 64;

  v8f acc[2][2] = {};
  const int ar = tid >> 3;            // 0..31  (row group)
  const int ak = (tid & 7) * 4;       // k quad
  const int bn = tid >> 2;            // 0..63  (d row)
  const int bq = (tid & 3) * 4;       // k quad base

  for (int kt = 0; kt < CIN / 32; ++kt) {
    const int c0 = kt * 32;
    if (kt + 1 < CIN / 32) {           // hint next tiles into cache
      __builtin_prefetch(&W_add[((size_t)ksub * DH + (n0 + bn)) * CIN + c0 + 32], 0, 1);
    }
    // A block: 128x32 f32 -> hi/lo bf16, swizzled
#pragma unroll
    for (int p = 0; p < 4; ++p) {
      int r = ar + p * 32;
      int m = m0 + r;
      int g = (m < BATCH) ? (ksub * BATCH + m) : (subj * BATCH + (m - BATCH));
      const float4 v = *(const float4*)&x[(size_t)g * CIN + c0 + ak];
      const float vv[4] = {v.x, v.y, v.z, v.w};
      int tm = r >> 4, mm = r & 15;
#pragma unroll
      for (int q = 0; q < 4; ++q) {
        unsigned short h = f2bf(vv[q]);
        unsigned short l = f2bf(vv[q] - bf2f(h));
        int idx = tm * 512 + a_frag_idx(mm, ak + q);
        As[0][idx] = h; As[1][idx] = l;
      }
    }
    // B block: W[ksub][n0+nn][c0..c0+31] 64x32
#pragma unroll
    for (int p = 0; p < 2; ++p) {
      int kkb = bq + p * 16;
      const float4 v = *(const float4*)&W_add[((size_t)ksub * DH + (n0 + bn)) * CIN + c0 + kkb];
      const float vv[4] = {v.x, v.y, v.z, v.w};
      int tn = bn >> 4, nloc = bn & 15;
#pragma unroll
      for (int q = 0; q < 4; ++q) {
        unsigned short h = f2bf(vv[q]);
        unsigned short l = f2bf(vv[q] - bf2f(h));
        int idx = tn * 512 + b_frag_idx(kkb + q, nloc);
        Bs[0][idx] = h; Bs[1][idx] = l;
      }
    }
    __syncthreads();
    v16bf ah[2], al[2], bh[2], bl[2];
#pragma unroll
    for (int i = 0; i < 2; ++i) {
      int tm = waveM * 2 + i;
      ah[i] = *reinterpret_cast<const v16bf*>(&As[0][tm * 512 + lane * 16]);
      al[i] = *reinterpret_cast<const v16bf*>(&As[1][tm * 512 + lane * 16]);
    }
#pragma unroll
    for (int j = 0; j < 2; ++j) {
      int tn = waveN * 2 + j;
      bh[j] = *reinterpret_cast<const v16bf*>(&Bs[0][tn * 512 + lane * 16]);
      bl[j] = *reinterpret_cast<const v16bf*>(&Bs[1][tn * 512 + lane * 16]);
    }
#pragma unroll
    for (int i = 0; i < 2; ++i)
#pragma unroll
      for (int j = 0; j < 2; ++j) {
        acc[i][j] = __builtin_amdgcn_wmma_f32_16x16x32_bf16(false, ah[i], false, bh[j], (short)0, acc[i][j], false, false);
        acc[i][j] = __builtin_amdgcn_wmma_f32_16x16x32_bf16(false, ah[i], false, bl[j], (short)0, acc[i][j], false, false);
        acc[i][j] = __builtin_amdgcn_wmma_f32_16x16x32_bf16(false, al[i], false, bh[j], (short)0, acc[i][j], false, false);
      }
    __syncthreads();
  }
  // epilogue: + bias, store hi/lo bf16 copy of total
  const int mlane = (lane >> 4) << 3;   // 0 or 8
  const int nlane = lane & 15;
#pragma unroll
  for (int i = 0; i < 2; ++i)
#pragma unroll
    for (int j = 0; j < 2; ++j) {
      int d = n0 + waveN * 32 + j * 16 + nlane;
      float bias = b_add[ksub * DH + d];
#pragma unroll
      for (int r = 0; r < 8; ++r) {
        int m = m0 + waveM * 32 + i * 16 + mlane + r;
        float val = acc[i][j][r] + bias;
        unsigned short h = f2bf(val);
        unsigned short l = f2bf(val - bf2f(h));
        size_t o = ((size_t)k * M2 + m) * DH + d;
        tot_hi[o] = h; tot_lo[o] = l;
      }
    }
}

// ---------- row norms sq[k][i] ----------
__global__ __launch_bounds__(256) void k_rowstats(
    const unsigned short* __restrict__ tot_hi, const unsigned short* __restrict__ tot_lo,
    float* __restrict__ sq) {
  int row = blockIdx.x * 8 + (threadIdx.x >> 5);
  int lane = threadIdx.x & 31;
  size_t base = (size_t)row * DH;
  float s = 0.f;
  for (int d = lane; d < DH; d += 32) {
    float v = bf2f(tot_hi[base + d]) + bf2f(tot_lo[base + d]);
    s += v * v;
  }
  s = wave_sum(s);
  if (lane == 0) sq[row] = s;
}

// ---------- column sums (for exact bandwidth) ----------
__global__ __launch_bounds__(256) void k_colsum(
    const unsigned short* __restrict__ tot_hi, const unsigned short* __restrict__ tot_lo,
    float* __restrict__ colsum) {
  int d = blockIdx.x * 256 + threadIdx.x;   // 0..511
  int k = blockIdx.z;
  int r0 = blockIdx.y * 256;
  float s = 0.f;
  for (int r = 0; r < 256; ++r) {
    size_t o = ((size_t)k * M2 + r0 + r) * DH + d;
    s += bf2f(tot_hi[o]) + bf2f(tot_lo[o]);
  }
  atomicAdd(&colsum[k * DH + d], s);
}

// ---------- bandwidth per k:  sum(d2) = 2m*sum(sq) - 2*||colsum||^2 ----------
__global__ __launch_bounds__(256) void k_bw(
    const float* __restrict__ sq, const float* __restrict__ colsum, float* __restrict__ bw) {
  int k = blockIdx.x;
  float s1 = 0.f;
  for (int i = threadIdx.x; i < M2; i += 256) s1 += sq[k * M2 + i];
  float s2 = 0.f;
  for (int d = threadIdx.x; d < DH; d += 256) { float c = colsum[k * DH + d]; s2 += c * c; }
  s1 = block_sum(s1);
  s2 = block_sum(s2);
  if (threadIdx.x == 0) {
    float sumd2 = 2.f * (float)M2 * s1 - 2.f * s2;
    float band  = sumd2 / (float)((size_t)M2 * M2 - M2);
    bw[k] = band * 0.25f;                 // / KERNEL_MUL^(KERNEL_NUM//2) = /4
  }
}

// ---------- gram + fused MMD kernel-sum (bf16x3 WMMA) ----------
__global__ __launch_bounds__(256) void k_gram_mmd(
    const unsigned short* __restrict__ tot_hi, const unsigned short* __restrict__ tot_lo,
    const float* __restrict__ sq, const float* __restrict__ bw,
    float* __restrict__ mmd_total) {
  __shared__ unsigned short As[2][128 * 32];
  __shared__ unsigned short Bs[2][64 * 32];
  const int tid = threadIdx.x, lane = tid & 31, wid = tid >> 5;
  const int waveM = wid >> 1, waveN = wid & 1;
  const int k = blockIdx.z;
  const int i0 = blockIdx.y * 128, j0 = blockIdx.x * 64;
  const size_t kb = (size_t)k * M2 * DH;

  v8f acc[2][2] = {};

  for (int kt = 0; kt < DH / 32; ++kt) {
    const int c0 = kt * 32;
    { // A: rows i0..i0+127, 32 cols (bf16 pairs -> uint)
      int r = tid >> 1;
      int h8 = (tid & 1) * 16;
      int tm = r >> 4, mm = r & 15;
      const unsigned* sh = (const unsigned*)&tot_hi[kb + (size_t)(i0 + r) * DH + c0 + h8];
      const unsigned* sl = (const unsigned*)&tot_lo[kb + (size_t)(i0 + r) * DH + c0 + h8];
#pragma unroll
      for (int u = 0; u < 8; ++u) {
        int kk = h8 + u * 2;
        int idx = tm * 512 + a_frag_idx(mm, kk);   // even kk -> pair contiguous
        *(unsigned*)&As[0][idx] = sh[u];
        *(unsigned*)&As[1][idx] = sl[u];
      }
    }
    { // B: rows j0..j0+63 become N-columns
      int r = tid & 63;
      int g = tid >> 6;                  // 0..3
      int tn = r >> 4, nloc = r & 15;
      int kkb = g * 8;
      const unsigned* sh = (const unsigned*)&tot_hi[kb + (size_t)(j0 + r) * DH + c0 + kkb];
      const unsigned* sl = (const unsigned*)&tot_lo[kb + (size_t)(j0 + r) * DH + c0 + kkb];
#pragma unroll
      for (int u = 0; u < 4; ++u) {
        int kk = kkb + u * 2;
        int idx = tn * 512 + b_frag_idx(kk, nloc);
        *(unsigned*)&Bs[0][idx] = sh[u];
        *(unsigned*)&Bs[1][idx] = sl[u];
      }
    }
    __syncthreads();
    v16bf ah[2], al[2], bh[2], bl[2];
#pragma unroll
    for (int i = 0; i < 2; ++i) {
      int tm = waveM * 2 + i;
      ah[i] = *reinterpret_cast<const v16bf*>(&As[0][tm * 512 + lane * 16]);
      al[i] = *reinterpret_cast<const v16bf*>(&As[1][tm * 512 + lane * 16]);
    }
#pragma unroll
    for (int j = 0; j < 2; ++j) {
      int tn = waveN * 2 + j;
      bh[j] = *reinterpret_cast<const v16bf*>(&Bs[0][tn * 512 + lane * 16]);
      bl[j] = *reinterpret_cast<const v16bf*>(&Bs[1][tn * 512 + lane * 16]);
    }
#pragma unroll
    for (int i = 0; i < 2; ++i)
#pragma unroll
      for (int j = 0; j < 2; ++j) {
        acc[i][j] = __builtin_amdgcn_wmma_f32_16x16x32_bf16(false, ah[i], false, bh[j], (short)0, acc[i][j], false, false);
        acc[i][j] = __builtin_amdgcn_wmma_f32_16x16x32_bf16(false, ah[i], false, bl[j], (short)0, acc[i][j], false, false);
        acc[i][j] = __builtin_amdgcn_wmma_f32_16x16x32_bf16(false, al[i], false, bh[j], (short)0, acc[i][j], false, false);
      }
    __syncthreads();
  }
  // fused epilogue: d2 -> 5 gaussians -> signed sum (XX+YY-XY-YX)
  const float bwk = bw[k];
  const float inv = 1.f / bwk;
  const int mlane = (lane >> 4) << 3;
  const int nlane = lane & 15;
  float local = 0.f;
#pragma unroll
  for (int i = 0; i < 2; ++i)
#pragma unroll
    for (int j = 0; j < 2; ++j) {
      int gj = j0 + waveN * 32 + j * 16 + nlane;
      float sqj = sq[k * M2 + gj];
#pragma unroll
      for (int r = 0; r < 8; ++r) {
        int gi = i0 + waveM * 32 + i * 16 + mlane + r;
        float d2 = sq[k * M2 + gi] + sqj - 2.f * acc[i][j][r];
        float t = d2 * inv;
        float ks = __expf(-t) + __expf(-0.5f * t) + __expf(-0.25f * t) +
                   __expf(-0.125f * t) + __expf(-0.0625f * t);
        float sgn = ((gi < BATCH) == (gj < BATCH)) ? 1.f : -1.f;
        local += sgn * ks;
      }
    }
  local = wave_sum(local);
  if (lane == 0) atomicAdd(mmd_total, local);
}

// ---------- P = softmax(outs, axis=-1) ----------
__global__ __launch_bounds__(256) void k_softmaxP(
    const unsigned short* __restrict__ tot_hi, const unsigned short* __restrict__ tot_lo,
    float* __restrict__ P) {
  int row = blockIdx.x * 8 + (threadIdx.x >> 5);   // 0..K*BATCH
  int lane = threadIdx.x & 31;
  int k = row >> 10, r = row & (BATCH - 1);
  size_t base = ((size_t)k * M2 + r) * DH;
  float v[16];
  float mx = -3.4e38f;
#pragma unroll
  for (int u = 0; u < 16; ++u) {
    int d = lane + u * 32;
    v[u] = bf2f(tot_hi[base + d]) + bf2f(tot_lo[base + d]);
    mx = fmaxf(mx, v[u]);
  }
#pragma unroll
  for (int o = 16; o > 0; o >>= 1) mx = fmaxf(mx, __shfl_xor(mx, o, 32));
  float s = 0.f;
#pragma unroll
  for (int u = 0; u < 16; ++u) { v[u] = __expf(v[u] - mx); s += v[u]; }
  s = wave_sum(s);
  float invs = 1.f / s;
#pragma unroll
  for (int u = 0; u < 16; ++u) P[(size_t)row * DH + lane + u * 32] = v[u] * invs;
}

// ---------- disc: sum over 91 pairs of |P_i - P_j| ----------
__global__ __launch_bounds__(256) void k_disc(const float* __restrict__ P, float* __restrict__ disc_acc) {
  int p = blockIdx.y;
  int i = 0, rem = p, cnt = KSRC - 1;
  while (rem >= cnt) { rem -= cnt; ++i; --cnt; }
  int j = i + 1 + rem;
  const float* Pi = P + (size_t)i * BATCH * DH;
  const float* Pj = P + (size_t)j * BATCH * DH;
  int base = blockIdx.x * 4096;
  float s = 0.f;
#pragma unroll
  for (int u = 0; u < 16; ++u) {
    int e = base + u * 256 + threadIdx.x;
    s += fabsf(Pi[e] - Pj[e]);
  }
  s = block_sum(s);
  if (threadIdx.x == 0) atomicAdd(disc_acc, s);
}

// ---------- classifier heads: cls KL + t_prob accumulation ----------
__global__ __launch_bounds__(256) void k_head(
    const unsigned short* __restrict__ tot_hi, const unsigned short* __restrict__ tot_lo,
    const float* __restrict__ W_cls, const float* __restrict__ b_cls,
    const float* __restrict__ y, const int* __restrict__ subjp,
    float* __restrict__ cls_acc, float* __restrict__ tprob) {
  int row = blockIdx.x * 8 + (threadIdx.x >> 5);   // 0..K*M2
  int lane = threadIdx.x & 31;
  int k = row >> 11, r = row & (M2 - 1);
  int subj = subjp[0];
  size_t base = (size_t)row * DH;
  const float* w0 = W_cls + (size_t)k * NCLS * DH;
  const float* w1 = w0 + DH;
  const float* w2 = w1 + DH;
  float a0 = 0.f, a1 = 0.f, a2 = 0.f;
  for (int d = lane; d < DH; d += 32) {
    float h = bf2f(tot_hi[base + d]) + bf2f(tot_lo[base + d]);
    h = fmaxf(h, 0.f);
    a0 += h * w0[d]; a1 += h * w1[d]; a2 += h * w2[d];
  }
  a0 = wave_sum(a0); a1 = wave_sum(a1); a2 = wave_sum(a2);
  if (lane == 0) {
    float l0 = a0 + b_cls[k * NCLS + 0];
    float l1 = a1 + b_cls[k * NCLS + 1];
    float l2 = a2 + b_cls[k * NCLS + 2];
    float mx = fmaxf(l0, fmaxf(l1, l2));
    float e0 = __expf(l0 - mx), e1 = __expf(l1 - mx), e2 = __expf(l2 - mx);
    float se = e0 + e1 + e2;
    if (r < BATCH) {                         // source rows -> KL loss
      int ksub = k + (k >= subj);
      const float* yr = y + ((size_t)ksub * BATCH + r) * NCLS;
      float lse = mx + __logf(se);
      float contrib = yr[0] * (lse - l0) + yr[1] * (lse - l1) + yr[2] * (lse - l2);
      atomicAdd(cls_acc, contrib);
    } else {                                 // target rows -> mean softmax (scale-free for argmax)
      int b = r - BATCH;
      float invs = 1.f / se;
      atomicAdd(&tprob[b * NCLS + 0], e0 * invs);
      atomicAdd(&tprob[b * NCLS + 1], e1 * invs);
      atomicAdd(&tprob[b * NCLS + 2], e2 * invs);
    }
  }
}

// ---------- prediction count ----------
__global__ void k_pred(const float* __restrict__ tprob, const float* __restrict__ y,
                       const int* __restrict__ subjp, int* __restrict__ pred) {
  int b = blockIdx.x * 256 + threadIdx.x;
  if (b >= BATCH) return;
  int subj = subjp[0];
  const float* t = &tprob[b * NCLS];
  int ai = 0; float av = t[0];
  if (t[1] > av) { av = t[1]; ai = 1; }
  if (t[2] > av) { av = t[2]; ai = 2; }
  const float* yt = &y[((size_t)subj * BATCH + b) * NCLS];
  int yi = 0; float yv = yt[0];
  if (yt[1] > yv) { yv = yt[1]; yi = 1; }
  if (yt[2] > yv) { yv = yt[2]; yi = 2; }
  if (ai == yi) atomicAdd(pred, 1);
}

// ---------- finalize 5 outputs ----------
__global__ void k_final(const float* mmd_total, const float* disc_acc,
                        const float* cls_acc, const int* pred, float* out) {
  if (threadIdx.x == 0 && blockIdx.x == 0) {
    out[0] = 0.1f * (*mmd_total) / ((float)KSRC * (float)BATCH * (float)BATCH);
    out[1] = 0.4f * (2.f / (float)(KSRC * (KSRC - 1))) * (*disc_acc) / ((float)BATCH * (float)DH);
    out[2] = (*cls_acc) / ((float)(KSRC * BATCH));
    out[3] = (float)(*pred);
    out[4] = (float)BATCH;
  }
}

extern "C" void kernel_launch(void* const* d_in, const int* in_sizes, int n_in,
                              void* d_out, int out_size, void* d_ws, size_t ws_size,
                              hipStream_t stream) {
  (void)in_sizes; (void)n_in; (void)out_size; (void)ws_size;
  const float* x     = (const float*)d_in[0];
  const float* y     = (const float*)d_in[1];
  const float* W_add = (const float*)d_in[3];
  const float* b_add = (const float*)d_in[4];
  const float* W_cls = (const float*)d_in[5];
  const float* b_cls = (const float*)d_in[6];
  const int*   subj  = (const int*)d_in[7];   // low 32 bits valid for int32 or int64 scalar

  char* ws = (char*)d_ws;
  size_t off = 0;
  auto take = [&](size_t bytes) -> void* {
    void* p = ws + off;
    off += (bytes + 255) & ~(size_t)255;
    return p;
  };
  unsigned short* tot_hi = (unsigned short*)take((size_t)KSRC * M2 * DH * 2);
  unsigned short* tot_lo = (unsigned short*)take((size_t)KSRC * M2 * DH * 2);
  float* P         = (float*)take((size_t)KSRC * BATCH * DH * 4);
  float* sq        = (float*)take((size_t)KSRC * M2 * 4);
  float* colsum    = (float*)take((size_t)KSRC * DH * 4);
  float* bw        = (float*)take((size_t)KSRC * 4);
  float* tprob     = (float*)take((size_t)BATCH * NCLS * 4);
  float* mmd_total = (float*)take(4);
  float* disc_acc  = (float*)take(4);
  float* cls_acc   = (float*)take(4);
  int*   pred      = (int*)take(4);

  k_zero<<<28, 256, 0, stream>>>(tprob, colsum, mmd_total, disc_acc, cls_acc, pred);
  k_gemm1<<<dim3(DH / 64, M2 / 128, KSRC), 256, 0, stream>>>(x, W_add, b_add, subj, tot_hi, tot_lo);
  k_rowstats<<<KSRC * M2 / 8, 256, 0, stream>>>(tot_hi, tot_lo, sq);
  k_colsum<<<dim3(DH / 256, M2 / 256, KSRC), 256, 0, stream>>>(tot_hi, tot_lo, colsum);
  k_bw<<<KSRC, 256, 0, stream>>>(sq, colsum, bw);
  k_gram_mmd<<<dim3(M2 / 64, M2 / 128, KSRC), 256, 0, stream>>>(tot_hi, tot_lo, sq, bw, mmd_total);
  k_softmaxP<<<KSRC * BATCH / 8, 256, 0, stream>>>(tot_hi, tot_lo, P);
  k_disc<<<dim3(BATCH * DH / 4096, KSRC * (KSRC - 1) / 2), 256, 0, stream>>>(P, disc_acc);
  k_head<<<KSRC * M2 / 8, 256, 0, stream>>>(tot_hi, tot_lo, W_cls, b_cls, y, subj, cls_acc, tprob);
  k_pred<<<(BATCH + 255) / 256, 256, 0, stream>>>(tprob, y, subj, pred);
  k_final<<<1, 32, 0, stream>>>(mmd_total, disc_acc, cls_acc, pred, (float*)d_out);
}
